// VideoEncoder_89842125898045
// MI455X (gfx1250) — matrix operand order repaired and checked by reading, not compiled, not run on hardware
//
#include <hip/hip_runtime.h>
#include <hip/hip_bf16.h>

typedef __bf16        v16bf __attribute__((ext_vector_type(16)));
typedef float         v8f   __attribute__((ext_vector_type(8)));
typedef unsigned int  u32x4 __attribute__((ext_vector_type(4)));
typedef float         f32x4 __attribute__((ext_vector_type(4)));

#define BB 8
#define TT 16
#define HH 256
#define WW 512
#define EMB 512
#define SEQ (TT*HH)     /* 4096 steps per batch row */
#define MTOT (BB*SEQ)   /* 32768 rows of seq        */
#define NGATE (4*EMB)   /* 2048 gate columns        */

// ---------------- WMMA fragment helpers (bf16 16x16x32) ----------------
// A (16x32, row-major): lane<16 -> M=lane, K chunks {k..k+7, k+16..k+23}
//                       lane>=16 -> M=lane-16, chunks shifted by +8.
// B (32x16 from w[N][K] row-major): lane<16 -> N=lane, K chunks {k..k+7, k+16..k+23};
//                                   lane>=16 -> N=lane-16, chunks shifted by +8.
struct FragBF { union { u32x4 q[2]; v16bf v; }; };

__device__ __forceinline__ void load_a(FragBF& f, const __hip_bfloat16* p) {
  const u32x4* q = (const u32x4*)p;
  f.q[0] = q[0];
  f.q[1] = q[2];          // +16 bf16 elements (=32 bytes)
}
__device__ __forceinline__ void load_b(FragBF& f, const __hip_bfloat16* p) {
  const u32x4* q = (const u32x4*)p;
  f.q[0] = q[0];
  f.q[1] = q[2];          // +16 bf16 elements (=32 bytes)
}

__device__ __forceinline__ float sigmoidf_(float x) { return 1.f / (1.f + __expf(-x)); }

// ---------------- weight pack + bias fold ----------------
__global__ void pack_kernel(const float* __restrict__ w_ih, const float* __restrict__ proj_w,
                            const float* __restrict__ b_ih, const float* __restrict__ b_hh,
                            __hip_bfloat16* __restrict__ w_ih_bf,
                            __hip_bfloat16* __restrict__ proj_w_bf,
                            float* __restrict__ bias_sum) {
  int i = blockIdx.x * 256 + threadIdx.x;
  if (i < NGATE * EMB) w_ih_bf[i]  = __float2bfloat16(w_ih[i]);
  if (i < EMB * EMB)   proj_w_bf[i] = __float2bfloat16(proj_w[i]);
  if (i < NGATE)       bias_sum[i] = b_ih[i] + b_hh[i];
}

// ---------------- fused conv1 -> conv2 -> leaky, emit bf16 seq ----------------
__global__ void conv_kernel(const float* __restrict__ img,
                            const float* __restrict__ c1w, const float* __restrict__ c1b,
                            const float* __restrict__ c2w, const float* __restrict__ c2b,
                            __hip_bfloat16* __restrict__ seq) {
  __shared__ float tin[36][37];
  __shared__ float tmid[34][35];
  const int frame = blockIdx.z;                 // b*16 + t
  const int x0 = blockIdx.x * 32, y0 = blockIdx.y * 32;
  const float* im = img + (size_t)frame * HH * WW;
  float w1[9], w2[9];
#pragma unroll
  for (int i = 0; i < 9; ++i) { w1[i] = c1w[i]; w2[i] = c2w[i]; }
  const float b1 = c1b[0], b2 = c2b[0];
  const int tid = threadIdx.x;

  for (int i = tid; i < 36 * 36; i += 256) {
    int yy = i / 36, xx = i % 36;
    int gy = y0 - 2 + yy, gx = x0 - 2 + xx;
    float v = 0.f;
    if (gy >= 0 && gy < HH && gx >= 0 && gx < WW) v = im[gy * WW + gx];
    tin[yy][xx] = v;
  }
  __syncthreads();
  for (int i = tid; i < 34 * 34; i += 256) {
    int yy = i / 34, xx = i % 34;
    float v = b1;
#pragma unroll
    for (int dy = 0; dy < 3; ++dy)
#pragma unroll
      for (int dx = 0; dx < 3; ++dx) v += tin[yy + dy][xx + dx] * w1[dy * 3 + dx];
    tmid[yy][xx] = v;
  }
  __syncthreads();
  for (int i = tid; i < 32 * 32; i += 256) {
    int y = i >> 5, x = i & 31;
    float v = b2;
#pragma unroll
    for (int dy = 0; dy < 3; ++dy)
#pragma unroll
      for (int dx = 0; dx < 3; ++dx) v += tmid[y + dy][x + dx] * w2[dy * 3 + dx];
    v = v > 0.f ? v : 0.01f * v;                        // leaky_relu(0.01)
    const int bb = frame >> 4, t = frame & 15;
    size_t m = (size_t)bb * SEQ + t * HH + (y0 + y);
    seq[m * WW + (x0 + x)] = __float2bfloat16(v);
  }
}

// ---------------- generic C[M,N] = A[M,512] x W[N,512]^T + bias ----------------
template <bool OUT_BF16>
__global__ void gemm_wmma(const __hip_bfloat16* __restrict__ A,
                          const __hip_bfloat16* __restrict__ Wt,
                          const float* __restrict__ bias, void* __restrict__ Cout, int N) {
  const int lane = threadIdx.x & 31;
  const int wave = threadIdx.x >> 5;
  const int mt = blockIdx.x;
  const int nt = blockIdx.y * 8 + wave;
  const int l15 = lane & 15;
  const int hi = lane >> 4;
  const __hip_bfloat16* arow = A + (size_t)(mt * 16 + l15) * EMB + hi * 8;
  const __hip_bfloat16* brow = Wt + (size_t)(nt * 16 + l15) * EMB + hi * 8;
  v8f acc = {};
#pragma unroll
  for (int k = 0; k < EMB; k += 32) {
    FragBF a, b;
    load_a(a, arow + k);
    load_b(b, brow + k);
    acc = __builtin_amdgcn_wmma_f32_16x16x32_bf16(false, a.v, false, b.v,
                                                  (short)0, acc, false, false);
  }
  const int col = nt * 16 + l15;
  const float bv = bias[col];
#pragma unroll
  for (int r = 0; r < 8; ++r) {
    int row = mt * 16 + r + hi * 8;                     // D: lanes>=16 carry M+8
    float v = acc[r] + bv;
    if (OUT_BF16) ((__hip_bfloat16*)Cout)[(size_t)row * N + col] = __float2bfloat16(v);
    else          ((float*)Cout)[(size_t)row * N + col] = v;
  }
}

// ---------------- persistent LSTM scan: 8 WGs, LDS-resident w_hh slices ------
#define NWG 8
#define SLICE 64      // h-elements owned per WG
#define NCOLS 256     // 4 gates x SLICE columns per WG
#define KPAD 520      // EMB + 8 bf16 pad -> breaks 16-way LDS bank conflict
#define GSTR 268      // gate-stage row stride (floats), conflict-free
#define HSTR 520      // staged-h row stride (bf16), conflict-free

__global__ void init_kernel(__hip_bfloat16* __restrict__ h_pad, unsigned* __restrict__ sync) {
  int i = blockIdx.x * 256 + threadIdx.x;
  if (i < 16 * EMB) h_pad[i] = __float2bfloat16(0.f);   // rows 8..15 stay zero pad
  if (i == 0) sync[0] = 0u;
}

__global__ void scan_kernel(const float* __restrict__ w_hh,
                            const __hip_bfloat16* __restrict__ xg,
                            __hip_bfloat16* __restrict__ h_pad,
                            __hip_bfloat16* __restrict__ hs_last,
                            unsigned* __restrict__ sync) {
  extern __shared__ char smem[];
  __hip_bfloat16* sW = (__hip_bfloat16*)smem;                         // [NCOLS][KPAD]
  float* sG = (float*)(smem + (size_t)NCOLS * KPAD * 2);              // [16][GSTR]
  __hip_bfloat16* sH = (__hip_bfloat16*)(smem + (size_t)NCOLS * KPAD * 2
                                              + (size_t)16 * GSTR * 4); // [16][HSTR]

  const int wg = blockIdx.x;          // owns h-elements [wg*64, wg*64+64)
  const int tid = threadIdx.x;        // 512 threads = 16 waves
  const int lane = tid & 31, wave = tid >> 5;
  const int l15 = lane & 15, hi = lane >> 4;

  // Stage this WG's 256 gate columns of w_hh into LDS (f32 -> bf16), column-major:
  // column c = q*SLICE + e  <->  w_hh row (q*EMB + wg*SLICE + e), 512 K contiguous.
  for (int i = tid; i < NCOLS * (EMB / 4); i += 512) {
    int c = i >> 7;                   // EMB/4 = 128 float4 per column
    int k4 = (i & 127) << 2;
    int q = c >> 6, el = c & 63;
    f32x4 v = *(const f32x4*)(w_hh + (size_t)(q * EMB + wg * SLICE + el) * EMB + k4);
    __hip_bfloat16* d = sW + (size_t)c * KPAD + k4;
    d[0] = __float2bfloat16(v.x); d[1] = __float2bfloat16(v.y);
    d[2] = __float2bfloat16(v.z); d[3] = __float2bfloat16(v.w);
  }

  // One LSTM cell (batch b0, element e0) per thread; c-state lives in a
  // register for the entire 4096-step scan.
  const int b0 = tid >> 6, e0 = tid & 63;
  float c0 = 0.f;

  __syncthreads();

  const int nt = wave;                // each of the 16 waves owns one N-tile
  const __hip_bfloat16* bp = sW + (size_t)(nt * 16 + l15) * KPAD + hi * 8;
  const __hip_bfloat16* ap = sH + (size_t)l15 * HSTR + hi * 8;

  for (int s = 0; s < SEQ; ++s) {
    // ---- cooperative stage: h_pad[16x512] (global, just synced) -> sH (LDS)
    {
      const u32x4* src = (const u32x4*)(h_pad + tid * 16);      // 32B per thread
      int row = tid >> 5, col = (tid & 31) * 16;
      u32x4* dst = (u32x4*)(sH + row * HSTR + col);
      dst[0] = src[0];
      dst[1] = src[1];
    }
    __syncthreads();

    // ---- gates_slice = sH[16x512] @ wslice^T (all fragments from LDS)
    v8f acc = {};
#pragma unroll 4
    for (int k = 0; k < EMB; k += 32) {
      FragBF a, b;
      load_a(a, ap + k);
      load_b(b, bp + k);
      acc = __builtin_amdgcn_wmma_f32_16x16x32_bf16(false, a.v, false, b.v,
                                                    (short)0, acc, false, false);
    }
#pragma unroll
    for (int r = 0; r < 8; ++r)
      sG[(r + hi * 8) * GSTR + nt * 16 + l15] = acc[r];
    __syncthreads();

    // ---- elementwise LSTM cell (torch gate order i,f,g,o)
    const size_t xb = ((size_t)(b0 * SEQ + s)) * NGATE + wg * SLICE + e0;
    {
      float gi = sG[b0 * GSTR + e0]        + __bfloat162float(xg[xb]);
      float gf = sG[b0 * GSTR + 64 + e0]   + __bfloat162float(xg[xb + EMB]);
      float gg = sG[b0 * GSTR + 128 + e0]  + __bfloat162float(xg[xb + 2 * EMB]);
      float go = sG[b0 * GSTR + 192 + e0]  + __bfloat162float(xg[xb + 3 * EMB]);
      c0 = sigmoidf_(gf) * c0 + sigmoidf_(gi) * tanhf(gg);
      float h = sigmoidf_(go) * tanhf(c0);
      h_pad[b0 * EMB + wg * SLICE + e0] = __float2bfloat16(h);
      if (s >= SEQ - HH)
        hs_last[((size_t)b0 * HH + (s - (SEQ - HH))) * EMB + wg * SLICE + e0] =
            __float2bfloat16(h);
    }

    // ---- prefetch next step's xg gate values (overlaps the barrier spin);
    //      lowers to global_prefetch_b8.
    if (s + 1 < SEQ) {
      __builtin_prefetch(xg + xb + NGATE, 0, 3);
      __builtin_prefetch(xg + xb + NGATE + EMB, 0, 3);
      __builtin_prefetch(xg + xb + NGATE + 2 * EMB, 0, 3);
      __builtin_prefetch(xg + xb + NGATE + 3 * EMB, 0, 3);
    }

    // ---- device-wide barrier across the 8 co-resident WGs.
    // Every thread releases its own h stores to device scope first.
    __threadfence();
    __syncthreads();
    if (tid == 0) {
      atomicAdd(sync, 1u);
      const unsigned target = (unsigned)(s + 1) * NWG;
      while (__hip_atomic_load(sync, __ATOMIC_RELAXED, __HIP_MEMORY_SCOPE_AGENT) < target)
        __builtin_amdgcn_s_sleep(1);
      __threadfence();    // acquire: invalidates the (read-only) WGP$ for the WGP
    }
    __syncthreads();
  }
}

// ---------------- host launcher ----------------
extern "C" void kernel_launch(void* const* d_in, const int* in_sizes, int n_in,
                              void* d_out, int out_size, void* d_ws, size_t ws_size,
                              hipStream_t stream) {
  (void)in_sizes; (void)n_in; (void)out_size; (void)ws_size;
  const float* images = (const float*)d_in[0];
  const float* c1w    = (const float*)d_in[1];
  const float* c1b    = (const float*)d_in[2];
  const float* c2w    = (const float*)d_in[3];
  const float* c2b    = (const float*)d_in[4];
  const float* w_ih   = (const float*)d_in[5];
  const float* w_hh   = (const float*)d_in[6];
  const float* b_ih   = (const float*)d_in[7];
  const float* b_hh   = (const float*)d_in[8];
  const float* proj_w = (const float*)d_in[9];
  const float* proj_b = (const float*)d_in[10];

  char* ws = (char*)d_ws;
  size_t off = 0;
  auto carve = [&](size_t bytes) -> void* {
    void* p = ws + off;
    off += (bytes + 255) & ~(size_t)255;
    return p;
  };
  __hip_bfloat16* seq       = (__hip_bfloat16*)carve((size_t)MTOT * EMB * 2);   //  32 MiB
  __hip_bfloat16* w_ih_bf   = (__hip_bfloat16*)carve((size_t)NGATE * EMB * 2);  //   2 MiB
  __hip_bfloat16* proj_w_bf = (__hip_bfloat16*)carve((size_t)EMB * EMB * 2);    // 512 KiB
  float*          bias_sum  = (float*)carve((size_t)NGATE * 4);
  __hip_bfloat16* xg        = (__hip_bfloat16*)carve((size_t)MTOT * NGATE * 2); // 128 MiB
  __hip_bfloat16* h_pad     = (__hip_bfloat16*)carve((size_t)16 * EMB * 2);
  __hip_bfloat16* hs_last   = (__hip_bfloat16*)carve((size_t)BB * HH * EMB * 2);//   2 MiB
  unsigned*       syncbuf   = (unsigned*)carve(256);

  pack_kernel<<<(NGATE * EMB) / 256, 256, 0, stream>>>(w_ih, proj_w, b_ih, b_hh,
                                                       w_ih_bf, proj_w_bf, bias_sum);
  conv_kernel<<<dim3(WW / 32, HH / 32, BB * TT), 256, 0, stream>>>(images, c1w, c1b,
                                                                   c2w, c2b, seq);
  gemm_wmma<true><<<dim3(MTOT / 16, NGATE / 128), 256, 0, stream>>>(seq, w_ih_bf,
                                                                    bias_sum, xg, NGATE);
  init_kernel<<<(16 * EMB + 255) / 256, 256, 0, stream>>>(h_pad, syncbuf);
  const size_t lds = (size_t)NCOLS * KPAD * 2 + (size_t)16 * GSTR * 4
                   + (size_t)16 * HSTR * 2;                                     // ~293 KiB
  scan_kernel<<<NWG, 512, lds, stream>>>(w_hh, xg, h_pad, hs_last, syncbuf);
  gemm_wmma<false><<<dim3((BB * HH) / 16, EMB / 128), 256, 0, stream>>>(hs_last, proj_w_bf,
                                                                        proj_b,
                                                                        (float*)d_out, EMB);
}
